// PLinear_82325933130247
// MI455X (gfx1250) — matrix operand order repaired and checked by reading, not compiled, not run on hardware
//
#include <hip/hip_runtime.h>

typedef __attribute__((ext_vector_type(16))) __bf16 bf16x16;
typedef __attribute__((ext_vector_type(8)))  __bf16 bf16x8;
typedef __attribute__((ext_vector_type(4)))  __bf16 bf16x4;
typedef __attribute__((ext_vector_type(8)))  float  f32x8;
typedef __attribute__((ext_vector_type(4)))  int    v4i_t;

#define M_DIM 16384
#define N_DIM 4096
#define K_DIM 4096

#define BM 128
#define BN 128
#define BK 64
#define LDS_STRIDE 72   // BK + 8 halves pad: 144B rows = 36 dwords -> conflict-free b128 frag reads

__device__ __forceinline__ unsigned short enc_w(float p, float n) {
    // (p>0) - (n>0) in {-1,0,1}: exact bf16 bit patterns
    bool bp = p > 0.0f, bn = n > 0.0f;
    return (bp == bn) ? (unsigned short)0u
                      : (bp ? (unsigned short)0x3F80u : (unsigned short)0xBF80u);
}

// CDNA5 async global->LDS copy (ASYNCcnt-tracked), 16B per lane.
__device__ __forceinline__ void async_copy_b128(const void* gsrc, void* ldst) {
    __builtin_amdgcn_global_load_async_to_lds_b128(
        (__attribute__((address_space(1))) v4i_t*)gsrc,
        (__attribute__((address_space(3))) v4i_t*)ldst, 0, 0);
}
__device__ __forceinline__ void wait_async0() {
    asm volatile("s_wait_asynccnt 0x0" ::: "memory");
}

// ---------------------------------------------------------------------------
// Prologue: w_eff = binarize(w_pos) - binarize(w_neg) as bf16 bits, [N,K]
// ---------------------------------------------------------------------------
__global__ __launch_bounds__(256) void binarize_weights(
    const float* __restrict__ wp, const float* __restrict__ wn,
    unsigned short* __restrict__ wb)
{
    int i = blockIdx.x * 256 + threadIdx.x;             // float4 index, exact grid
    const float4 p = reinterpret_cast<const float4*>(wp)[i];
    const float4 q = reinterpret_cast<const float4*>(wn)[i];
    ushort4 r;
    r.x = enc_w(p.x, q.x);
    r.y = enc_w(p.y, q.y);
    r.z = enc_w(p.z, q.z);
    r.w = enc_w(p.w, q.w);
    reinterpret_cast<ushort4*>(wb)[i] = r;
}

// ---------------------------------------------------------------------------
// GEMM: Y[M,N] = X[M,K] (f32, split to bf16 hi+lo on the fly) @ Wb[N,K]^T
// 256 threads = 8 waves; block tile 128x128; wave tile 32x64 (2x4 WMMA tiles)
// B tile staged via global_load_async_to_lds_b128 (no VGPR staging).
// ---------------------------------------------------------------------------
__global__ __launch_bounds__(256) void plinear_wmma_gemm(
    const float* __restrict__ X,
    const unsigned short* __restrict__ Wb,
    float* __restrict__ Y)
{
    __shared__ __align__(16) unsigned short Ahi[2][BM * LDS_STRIDE];
    __shared__ __align__(16) unsigned short Alo[2][BM * LDS_STRIDE];
    __shared__ __align__(16) unsigned short Bs [2][BN * LDS_STRIDE];

    const int tid   = threadIdx.x;
    const int lane  = tid & 31;
    const int wave  = tid >> 5;
    const int waveM = wave & 3;    // 4 waves along M
    const int waveN = wave >> 2;   // 2 waves along N
    const int bm = blockIdx.y * BM;
    const int bn = blockIdx.x * BN;

    float4 regA[8];   // staged f32 x-tile chunks (128x64 f32 = 2048 float4s)

    f32x8 acc[2][4];
    #pragma unroll
    for (int tm = 0; tm < 2; ++tm)
        #pragma unroll
        for (int tn = 0; tn < 4; ++tn)
            acc[tm][tn] = (f32x8)(0.0f);

    // A tile: prefetch f32 into registers (conversion happens at LDS store)
    auto gloadA = [&](int kt) {
        #pragma unroll
        for (int i = 0; i < 8; ++i) {
            int c = tid + 256 * i;
            int row = c >> 4, col = (c & 15) << 2;
            regA[i] = *reinterpret_cast<const float4*>(
                X + (size_t)(bm + row) * K_DIM + kt + col);
        }
    };

    // B tile: async copy straight into LDS (bf16, no conversion needed)
    auto asyncB = [&](int kt, int buf) {
        #pragma unroll
        for (int i = 0; i < 4; ++i) {
            int c = tid + 256 * i;
            int row = c >> 3, col = (c & 7) << 3;
            async_copy_b128(Wb + (size_t)(bn + row) * K_DIM + kt + col,
                            &Bs[buf][row * LDS_STRIDE + col]);
        }
    };

    // Split staged f32 A into bf16 hi/lo planes in LDS
    auto sstoreA = [&](int buf) {
        #pragma unroll
        for (int i = 0; i < 8; ++i) {
            int c = tid + 256 * i;
            int row = c >> 4, col = (c & 15) << 2;
            const float f[4] = {regA[i].x, regA[i].y, regA[i].z, regA[i].w};
            bf16x4 hv, lv;
            #pragma unroll
            for (int j = 0; j < 4; ++j) {
                __bf16 h = (__bf16)f[j];
                hv[j] = h;
                lv[j] = (__bf16)(f[j] - (float)h);
            }
            *reinterpret_cast<bf16x4*>(&Ahi[buf][row * LDS_STRIDE + col]) = hv;
            *reinterpret_cast<bf16x4*>(&Alo[buf][row * LDS_STRIDE + col]) = lv;
        }
    };

    auto compute = [&](int buf) {
        #pragma unroll
        for (int ks = 0; ks < BK; ks += 32) {
            // A fragments (16x32 bf16): lane<16 -> K {0..7,16..23}; lane>=16 -> K {8..15,24..31}
            bf16x16 a[4];  // [tm*2 + plane(hi/lo)]
            const int arow  = waveM * 32 + (lane & 15);
            const int akoff = ks + ((lane & 16) ? 8 : 0);
            #pragma unroll
            for (int tm = 0; tm < 2; ++tm) {
                const unsigned short* ph = &Ahi[buf][(arow + 16 * tm) * LDS_STRIDE + akoff];
                bf16x8 h0 = *reinterpret_cast<const bf16x8*>(ph);
                bf16x8 h1 = *reinterpret_cast<const bf16x8*>(ph + 16);
                a[tm * 2 + 0] = __builtin_shufflevector(h0, h1,
                    0,1,2,3,4,5,6,7,8,9,10,11,12,13,14,15);
                const unsigned short* pl = &Alo[buf][(arow + 16 * tm) * LDS_STRIDE + akoff];
                bf16x8 l0 = *reinterpret_cast<const bf16x8*>(pl);
                bf16x8 l1 = *reinterpret_cast<const bf16x8*>(pl + 16);
                a[tm * 2 + 1] = __builtin_shufflevector(l0, l1,
                    0,1,2,3,4,5,6,7,8,9,10,11,12,13,14,15);
            }
            // B fragments (32x16 bf16): lane<16 -> K {0..7,16..23}; lane>=16 -> K {8..15,24..31} of col n=lane%16
            bf16x16 b[4];
            const int brow  = waveN * 64 + (lane & 15);
            const int bkoff = ks + ((lane & 16) ? 8 : 0);
            #pragma unroll
            for (int tn = 0; tn < 4; ++tn) {
                const unsigned short* pb = &Bs[buf][(brow + 16 * tn) * LDS_STRIDE + bkoff];
                bf16x8 b0 = *reinterpret_cast<const bf16x8*>(pb);
                bf16x8 b1 = *reinterpret_cast<const bf16x8*>(pb + 16);
                b[tn] = __builtin_shufflevector(b0, b1,
                    0,1,2,3,4,5,6,7,8,9,10,11,12,13,14,15);
            }
            #pragma unroll
            for (int tm = 0; tm < 2; ++tm)
                #pragma unroll
                for (int tn = 0; tn < 4; ++tn) {
                    acc[tm][tn] = __builtin_amdgcn_wmma_f32_16x16x32_bf16(
                        false, a[tm * 2 + 0], false, b[tn],
                        (short)0, acc[tm][tn], false, false);
                    acc[tm][tn] = __builtin_amdgcn_wmma_f32_16x16x32_bf16(
                        false, a[tm * 2 + 1], false, b[tn],
                        (short)0, acc[tm][tn], false, false);
                }
        }
    };

    gloadA(0);
    asyncB(0, 0);
    sstoreA(0);
    wait_async0();
    __syncthreads();
    int buf = 0;
    for (int kt = BK; kt < K_DIM; kt += BK) {
        gloadA(kt);            // prefetch next A into regs
        asyncB(kt, buf ^ 1);   // async next B straight into LDS (safe: all waves
                               // finished reading buf^1 at the previous barrier)
        compute(buf);
        sstoreA(buf ^ 1);
        wait_async0();         // own async B writes complete
        __syncthreads();
        buf ^= 1;
    }
    compute(buf);

    // C/D layout: VGPR r -> M = r (lanes 0-15) or r+8 (lanes 16-31), N = lane%16
    #pragma unroll
    for (int tm = 0; tm < 2; ++tm)
        #pragma unroll
        for (int tn = 0; tn < 4; ++tn) {
            const int row0 = bm + waveM * 32 + 16 * tm + ((lane & 16) ? 8 : 0);
            const int col  = bn + waveN * 64 + 16 * tn + (lane & 15);
            #pragma unroll
            for (int r = 0; r < 8; ++r)
                Y[(size_t)(row0 + r) * N_DIM + col] = acc[tm][tn][r];
        }
}

// ---------------------------------------------------------------------------
extern "C" void kernel_launch(void* const* d_in, const int* in_sizes, int n_in,
                              void* d_out, int out_size, void* d_ws, size_t ws_size,
                              hipStream_t stream) {
    const float* x  = (const float*)d_in[0];
    const float* wp = (const float*)d_in[1];
    const float* wn = (const float*)d_in[2];
    float* y = (float*)d_out;
    unsigned short* wb = (unsigned short*)d_ws;   // 4096*4096*2B = 32 MB

    // binarize weights: N*K/4 float4 elements, exact multiple of 256
    binarize_weights<<<dim3((N_DIM * (size_t)K_DIM / 4) / 256), 256, 0, stream>>>(wp, wn, wb);

    dim3 grid(N_DIM / BN, M_DIM / BM);   // (32, 128)
    plinear_wmma_gemm<<<grid, 256, 0, stream>>>(x, wb, y);
}
